// Dense_SGC_22170621182126
// MI455X (gfx1250) — matrix operand order repaired and checked by reading, not compiled, not run on hardware
//
#include <hip/hip_runtime.h>
#include <hip/hip_bf16.h>

// ---------------------------------------------------------------------------
// Dense_SGC pipeline for MI455X (gfx1250, wave32, WMMA).
//   B=8 N=8192 E=262144 F_IN=64 H=128 F_OUT=32, rows = B*N = 65536.
// Dense GEMMs: bf16 WMMA (v_wmma_f32_16x16x32_bf16, f32 accum), padded-LDS
// tiles (no bank conflicts), ks-outer/nt-inner schedule (independent WMMAs
// back-to-back), LDS-staged coalesced b128 output stores.
// BatchNorms folded into following GEMM weights; W02@Wc chained on device.
// Sparse aggregation via coalesced f32 atomics (one wave per edge).
// ---------------------------------------------------------------------------

typedef __attribute__((ext_vector_type(16))) __bf16        v16bf;
typedef __attribute__((ext_vector_type(8)))  float         v8f;
typedef __attribute__((ext_vector_type(4)))  unsigned int  v4u;

union FragU { v16bf v; v4u q[2]; };

static __device__ __forceinline__ unsigned short f2bf(float f) {
    unsigned int u = __float_as_uint(f);
    u += 0x7FFFu + ((u >> 16) & 1u);           // round-to-nearest-even
    return (unsigned short)(u >> 16);
}
static __device__ __forceinline__ float bf2f(unsigned short s) {
    return __uint_as_float(((unsigned int)s) << 16);
}

// A fragment (16x32 bf16, MxK), LDS row stride = LD ushorts.
// Lane L: row m=L&15, half=L>>4; element e -> K = ((e>>3)<<4) + half*8 + (e&7)
// => two contiguous 16B chunks at +0 and +32B.
template<int LD>
static __device__ __forceinline__ v16bf load_afrag(const unsigned short* base, int lane) {
    int m = lane & 15, half = (lane >> 4) & 1;
    const unsigned short* p = base + m * LD + half * 8;
    FragU f;
    f.q[0] = *(const v4u*)(p);
    f.q[1] = *(const v4u*)(p + 16);
    return f.v;
}

// B fragment (32x16 bf16, KxN) from LDS panel stored [N][K], stride LD.
// Lane L: col n=L&15, half=L>>4; element e -> K = ((e>>3)<<4) + half*8 + (e&7).
template<int LD>
static __device__ __forceinline__ v16bf load_bfrag(const unsigned short* base, int lane) {
    int n = lane & 15, half = (lane >> 4) & 1;
    const unsigned short* p = base + n * LD + half * 8;
    FragU f;
    f.q[0] = *(const v4u*)(p);
    f.q[1] = *(const v4u*)(p + 16);
    return f.v;
}

// ---------------------------------------------------------------------------
// Generic bf16 WMMA GEMM: out[rows x NOUT] = act(A[rows x K] @ W[K x NOUT] + bias)
// WT is bf16 transposed weights [NOUT][K]. 256 threads (8 waves), 128-row tile.
// ---------------------------------------------------------------------------
template<int K, int NOUT, bool RELU, bool OUTF32>
__global__ __launch_bounds__(256) void gemm_bf16_kernel(
    const unsigned short* __restrict__ A,
    const unsigned short* __restrict__ WT,
    const float* __restrict__ bias,
    void* __restrict__ outp)
{
    constexpr int LDA   = K + 8;                         // padded LDS stride
    constexpr int NT    = NOUT / 16;
    constexpr size_t STAGE = (size_t)(128 + NOUT) * LDA * 2;
    constexpr size_t OUTB  = (size_t)128 * NOUT * (OUTF32 ? 4 : 2);
    __shared__ __align__(16) char smem[(STAGE > OUTB ? STAGE : OUTB)];
    unsigned short* lA = (unsigned short*)smem;          // [128][LDA]
    unsigned short* lW = lA + 128 * LDA;                 // [NOUT][LDA]

    const int tid  = threadIdx.x;
    const int lane = tid & 31;
    const int wave = tid >> 5;
    const long long rowBase = (long long)blockIdx.x * 128;

    // Stage weight panel into padded LDS (16B vectors, coalesced).
#pragma unroll 2
    for (int i = tid; i < NOUT * (K / 8); i += 256) {
        int r = i / (K / 8), c = i % (K / 8);
        *(v4u*)&lW[r * LDA + c * 8] = *(const v4u*)&WT[r * K + c * 8];
    }
    // Stage A tile (128 x K bf16) into padded LDS.
    const unsigned short* Ab = A + rowBase * K;
#pragma unroll 2
    for (int i = tid; i < 128 * (K / 8); i += 256) {
        int r = i / (K / 8), c = i % (K / 8);
        *(v4u*)&lA[r * LDA + c * 8] = *(const v4u*)&Ab[r * K + c * 8];
    }
    __syncthreads();

    // Init accumulators with bias (broadcast along rows; n = lane&15).
    v8f acc[NT];
#pragma unroll
    for (int nt = 0; nt < NT; ++nt) {
        float bv = bias[nt * 16 + (lane & 15)];
#pragma unroll
        for (int i = 0; i < 8; ++i) acc[nt][i] = bv;
    }

    // ks-outer / nt-inner: consecutive WMMAs write different accumulators,
    // so the matrix pipe never stalls on RAW hazards.
    const unsigned short* aStrip = &lA[wave * 16 * LDA];
#pragma unroll
    for (int ks = 0; ks < K / 32; ++ks) {
        v16bf a = load_afrag<LDA>(aStrip + ks * 32, lane);
#pragma unroll
        for (int nt = 0; nt < NT; ++nt) {
            v16bf b = load_bfrag<LDA>(&lW[nt * 16 * LDA] + ks * 32, lane);
            acc[nt] = __builtin_amdgcn_wmma_f32_16x16x32_bf16(
                false, a, false, b, (short)0, acc[nt], false, false);
        }
    }

    // Stage results into LDS (reuse smem; accs live in registers), then do
    // fully coalesced vector stores to global.
    __syncthreads();
    const int n0 = lane & 15;
    const int mb = 8 * (lane >> 4);
    if (OUTF32) {
        float* oT = (float*)smem;                         // [128][NOUT]
#pragma unroll
        for (int nt = 0; nt < NT; ++nt) {
#pragma unroll
            for (int i = 0; i < 8; ++i) {
                float v = acc[nt][i];
                if (RELU) v = v > 0.f ? v : 0.f;
                oT[(wave * 16 + mb + i) * NOUT + nt * 16 + n0] = v;
            }
        }
        __syncthreads();
        float* og = (float*)outp + rowBase * NOUT;
#pragma unroll 2
        for (int i = tid * 4; i < 128 * NOUT; i += 256 * 4)
            *(float4*)&og[i] = *(float4*)&((float*)smem)[i];
    } else {
        unsigned short* oT = (unsigned short*)smem;       // [128][NOUT]
#pragma unroll
        for (int nt = 0; nt < NT; ++nt) {
#pragma unroll
            for (int i = 0; i < 8; ++i) {
                float v = acc[nt][i];
                if (RELU) v = v > 0.f ? v : 0.f;
                oT[(wave * 16 + mb + i) * NOUT + nt * 16 + n0] = f2bf(v);
            }
        }
        __syncthreads();
        unsigned short* og = (unsigned short*)outp + rowBase * NOUT;
#pragma unroll 2
        for (int i = tid * 8; i < 128 * NOUT; i += 256 * 8)
            *(v4u*)&og[i] = *(v4u*)&((unsigned short*)smem)[i];
    }
}

// ---------------------------------------------------------------------------
// f32 -> bf16 convert
// ---------------------------------------------------------------------------
__global__ __launch_bounds__(256) void f32_to_bf16_kernel(
    const float* __restrict__ in, unsigned short* __restrict__ out, long long n)
{
    long long i = (long long)blockIdx.x * 256 + threadIdx.x;
    long long stride = (long long)gridDim.x * 256;
    for (; i < n; i += stride) out[i] = f2bf(in[i]);
}

// ---------------------------------------------------------------------------
// Per-feature sum / sumsq over bf16 matrix [rows x 128] -> stats[0:128]=sum,
// stats[128:256]=sumsq (atomic accumulate; caller zeroes stats).
// ---------------------------------------------------------------------------
__global__ __launch_bounds__(256) void stats_kernel(
    const unsigned short* __restrict__ X, float* __restrict__ stats, int rows)
{
    __shared__ float ls[256];
    __shared__ float lss[256];
    int tid = threadIdx.x;
    int f = tid & 127;
    float s = 0.f, ss = 0.f;
    long long total = (long long)rows * 128;
    for (long long i = (long long)blockIdx.x * 256 + tid; i < total;
         i += (long long)gridDim.x * 256) {
        float v = bf2f(X[i]);
        s += v; ss += v * v;
    }
    ls[tid] = s; lss[tid] = ss;
    __syncthreads();
    if (tid < 128) {
        atomicAdd(&stats[f],        ls[tid] + ls[tid + 128]);
        atomicAdd(&stats[128 + f], lss[tid] + lss[tid + 128]);
    }
}

// ---------------------------------------------------------------------------
// Fold BN (from stats) into weight W[K x NOUT]: WT[j][k] = bf(s_k * W[k][j]),
// bias_out[j] = bias_in[j] + sum_k t_k W[k][j]. stats==null => identity BN.
// grid = K blocks, block = NOUT threads.
// ---------------------------------------------------------------------------
__global__ void fold_bn_weight_kernel(
    const float* __restrict__ stats, const float* __restrict__ bnw,
    const float* __restrict__ bnb, const float* __restrict__ W,
    const float* __restrict__ bias_in, unsigned short* __restrict__ WT,
    float* __restrict__ bias_out, int K, int NOUT, float invCnt)
{
    int k = blockIdx.x, j = threadIdx.x;
    float s = 1.f;
    if (stats) {
        float mean = stats[k] * invCnt;
        float var  = stats[K + k] * invCnt - mean * mean;
        s = bnw[k] * rsqrtf(var + 1e-5f);
    }
    WT[(long long)j * K + k] = f2bf(s * W[(long long)k * NOUT + j]);
    if (k == 0) {
        float acc = bias_in[j];
        if (stats) {
            for (int kk = 0; kk < K; ++kk) {
                float mean = stats[kk] * invCnt;
                float var  = stats[K + kk] * invCnt - mean * mean;
                float sk = bnw[kk] * rsqrtf(var + 1e-5f);
                float tk = bnb[kk] - mean * sk;
                acc += tk * W[(long long)kk * NOUT + j];
            }
        }
        bias_out[j] = acc;
    }
}

// ---------------------------------------------------------------------------
// Fold BN0 + (W02 @ Wc) into one 128x128 weight. WmT[j][k] = bf(s0_k*(W02@Wc)[k][j])
// grid = 128 blocks (k), block = 128 threads (j).
// ---------------------------------------------------------------------------
__global__ __launch_bounds__(128) void fold_mid_W_kernel(
    const float* __restrict__ stats0, const float* __restrict__ bn0w,
    const float* __restrict__ W02, const float* __restrict__ Wc,
    unsigned short* __restrict__ WmT, float invCnt)
{
    __shared__ float row[128];
    int k = blockIdx.x, j = threadIdx.x;
    row[j] = W02[k * 128 + j];
    __syncthreads();
    float acc = 0.f;
    for (int m = 0; m < 128; ++m) acc += row[m] * Wc[m * 128 + j];
    float mean = stats0[k] * invCnt;
    float var  = stats0[128 + k] * invCnt - mean * mean;
    float s = bn0w[k] * rsqrtf(var + 1e-5f);
    WmT[j * 128 + k] = f2bf(s * acc);
}

__global__ __launch_bounds__(128) void fold_mid_b_kernel(
    const float* __restrict__ stats0, const float* __restrict__ bn0w,
    const float* __restrict__ bn0b, const float* __restrict__ W02,
    const float* __restrict__ b02, const float* __restrict__ Wc,
    const float* __restrict__ bc, float* __restrict__ bm, float invCnt)
{
    __shared__ float t0[128];
    __shared__ float b02p[128];
    int j = threadIdx.x;
    {
        float mean = stats0[j] * invCnt;
        float var  = stats0[128 + j] * invCnt - mean * mean;
        float s = bn0w[j] * rsqrtf(var + 1e-5f);
        t0[j] = bn0b[j] - mean * s;
    }
    __syncthreads();
    float acc = b02[j];
    for (int k = 0; k < 128; ++k) acc += t0[k] * W02[k * 128 + j];
    b02p[j] = acc;
    __syncthreads();
    float acc2 = bc[j];
    for (int m = 0; m < 128; ++m) acc2 += b02p[m] * Wc[m * 128 + j];
    bm[j] = acc2;
}

// ---------------------------------------------------------------------------
// Edge scatter: one wave per edge, lane handles 4 consecutive features.
// agg[b,row,:] += val * hc[b,col,:]; deg[b,row] += val.
// ---------------------------------------------------------------------------
__global__ __launch_bounds__(256) void aggregate_kernel(
    const float* __restrict__ hc, const int* __restrict__ erows,
    const int* __restrict__ ecols, const float* __restrict__ evals,
    float* __restrict__ agg, float* __restrict__ deg)
{
    const long long E = 262144, Nn = 8192;
    long long g = ((long long)blockIdx.x * 256 + threadIdx.x) >> 5;  // edge id
    int lane = threadIdx.x & 31;
    if (g >= 8 * E) return;
    int b   = (int)(g / E);
    int row = erows[g];
    int col = ecols[g];
    float val = evals[g];
    const float4* src = (const float4*)&hc[((long long)b * Nn + col) * 128];
    float4 v = src[lane];
    float* dst = &agg[((long long)b * Nn + row) * 128 + lane * 4];
    atomicAdd(dst + 0, val * v.x);
    atomicAdd(dst + 1, val * v.y);
    atomicAdd(dst + 2, val * v.z);
    atomicAdd(dst + 3, val * v.w);
    if (lane == 0) atomicAdd(&deg[(long long)b * Nn + row], val);
}

// ---------------------------------------------------------------------------
// hr = bf16( relu(agg / max(deg,1)) * (n < num_nodes[b]) )
// ---------------------------------------------------------------------------
__global__ __launch_bounds__(256) void relu_norm_kernel(
    const float* __restrict__ agg, const float* __restrict__ deg,
    const int* __restrict__ num_nodes, unsigned short* __restrict__ hr)
{
    long long total = (long long)65536 * 128;
    long long stride = (long long)gridDim.x * 256;
    for (long long i = (long long)blockIdx.x * 256 + threadIdx.x; i < total; i += stride) {
        long long r = i >> 7;
        int b = (int)(r >> 13);
        int n = (int)(r & 8191);
        float d = deg[r];
        if (d < 1.f) d = 1.f;
        float v = agg[i] / d;
        v = v > 0.f ? v : 0.f;
        if (n >= num_nodes[b]) v = 0.f;
        hr[i] = f2bf(v);
    }
}

// ---------------------------------------------------------------------------
extern "C" void kernel_launch(void* const* d_in, const int* in_sizes, int n_in,
                              void* d_out, int out_size, void* d_ws, size_t ws_size,
                              hipStream_t stream) {
    (void)in_sizes; (void)n_in; (void)out_size; (void)ws_size;
    const long long ROWS = 65536;                 // B*N
    const float invCnt = 1.f / (float)ROWS;

    const float* x     = (const float*)d_in[0];
    const int*   erows = (const int*)d_in[1];
    const int*   ecols = (const int*)d_in[2];
    const float* evals = (const float*)d_in[3];
    const int*   nnod  = (const int*)d_in[4];
    const float* W0  = (const float*)d_in[5];   const float* b0  = (const float*)d_in[6];
    const float* W02 = (const float*)d_in[7];   const float* b02 = (const float*)d_in[8];
    const float* Wc  = (const float*)d_in[9];   const float* bc  = (const float*)d_in[10];
    const float* W1  = (const float*)d_in[11];  const float* b1  = (const float*)d_in[12];
    const float* W2  = (const float*)d_in[13];  const float* b2  = (const float*)d_in[14];
    const float* bn0w = (const float*)d_in[15]; const float* bn0b = (const float*)d_in[16];
    const float* bncw = (const float*)d_in[17]; const float* bncb = (const float*)d_in[18];
    const float* bn1w = (const float*)d_in[19]; const float* bn1b = (const float*)d_in[20];

    // Workspace carve-up (256B aligned).
    char* base = (char*)d_ws;
    size_t off = 0;
    auto carve = [&](size_t bytes) { char* p = base + off;
        off = (off + bytes + 255) & ~(size_t)255; return p; };
    unsigned short* xbf = (unsigned short*)carve(ROWS * 64 * 2);
    unsigned short* h0  = (unsigned short*)carve(ROWS * 128 * 2);
    float*          hc  = (float*)carve(ROWS * 128 * 4);
    float*          agg = (float*)carve(ROWS * 128 * 4);
    float*          deg = (float*)carve(ROWS * 4);
    unsigned short* hr  = (unsigned short*)carve(ROWS * 128 * 2);
    unsigned short* h3  = (unsigned short*)carve(ROWS * 128 * 2);
    float*          st  = (float*)carve(3 * 256 * 4);   // stats0 | statsc | stats1
    unsigned short* W0T = (unsigned short*)carve(128 * 64 * 2);
    unsigned short* WmT = (unsigned short*)carve(128 * 128 * 2);
    unsigned short* W1T = (unsigned short*)carve(128 * 128 * 2);
    unsigned short* W2T = (unsigned short*)carve(32 * 128 * 2);
    float* bias0 = (float*)carve(128 * 4);
    float* bm    = (float*)carve(128 * 4);
    float* b1p   = (float*)carve(128 * 4);
    float* b2p   = (float*)carve(32 * 4);
    float* stats0 = st, *statsc = st + 256, *stats1 = st + 512;

    // Zero the accumulated buffers (graph-capture-safe memset nodes).
    hipMemsetAsync(st, 0, 3 * 256 * 4, stream);
    hipMemsetAsync(agg, 0, (size_t)ROWS * 128 * 4, stream);
    hipMemsetAsync(deg, 0, (size_t)ROWS * 4, stream);

    // 1) x -> bf16
    f32_to_bf16_kernel<<<2048, 256, 0, stream>>>(x, xbf, ROWS * 64);
    // 2) W0 -> bf16 transposed (identity BN)
    fold_bn_weight_kernel<<<64, 128, 0, stream>>>(
        nullptr, nullptr, nullptr, W0, b0, W0T, bias0, 64, 128, 0.f);
    // 3) h0 = relu(x @ W0 + b0)   [bf16 out]
    gemm_bf16_kernel<64, 128, true, false><<<ROWS / 128, 256, 0, stream>>>(xbf, W0T, bias0, h0);
    // 4) BN0 stats over h0
    stats_kernel<<<256, 256, 0, stream>>>(h0, stats0, (int)ROWS);
    // 5) Fold BN0 + W02 + Wc into one weight/bias
    fold_mid_W_kernel<<<128, 128, 0, stream>>>(stats0, bn0w, W02, Wc, WmT, invCnt);
    fold_mid_b_kernel<<<1, 128, 0, stream>>>(stats0, bn0w, bn0b, W02, b02, Wc, bc, bm, invCnt);
    // 6) hc = BN0(h0) @ W02 @ Wc + bm   [f32 out for aggregation]
    gemm_bf16_kernel<128, 128, false, true><<<ROWS / 128, 256, 0, stream>>>(h0, WmT, bm, hc);
    // 7) Sparse aggregation (one wave per edge, 8 waves/block)
    aggregate_kernel<<<262144, 256, 0, stream>>>(hc, erows, ecols, evals, agg, deg);
    // 8) hr = relu(agg/deg) * mask  [bf16]
    relu_norm_kernel<<<4096, 256, 0, stream>>>(agg, deg, nnod, hr);
    // 9) BNc stats + fold into W1
    stats_kernel<<<256, 256, 0, stream>>>(hr, statsc, (int)ROWS);
    fold_bn_weight_kernel<<<128, 128, 0, stream>>>(statsc, bncw, bncb, W1, b1, W1T, b1p, 128, 128, invCnt);
    // 10) h3 = relu(BNc(hr) @ W1 + b1)  [bf16]
    gemm_bf16_kernel<128, 128, true, false><<<ROWS / 128, 256, 0, stream>>>(hr, W1T, b1p, h3);
    // 11) BN1 stats + fold into W2
    stats_kernel<<<256, 256, 0, stream>>>(h3, stats1, (int)ROWS);
    fold_bn_weight_kernel<<<128, 32, 0, stream>>>(stats1, bn1w, bn1b, W2, b2, W2T, b2p, 128, 32, invCnt);
    // 12) out = BN1(h3) @ W2 + b2   [f32, written to d_out]
    gemm_bf16_kernel<128, 32, false, true><<<ROWS / 128, 256, 0, stream>>>(h3, W2T, b2p, (float*)d_out);
}
